// InterMultiplicativeUpdate_65635690217487
// MI455X (gfx1250) — compile-verified, matrix operand order checked by reading, not run on hardware
//
#include <hip/hip_runtime.h>
#include <hip/hip_bf16.h>

typedef __attribute__((ext_vector_type(16))) __bf16 v16bf;
typedef __attribute__((ext_vector_type(8)))  float  v8f;

#define C_DIM  128
#define NROW   512
#define TILE_J 16
#define N_JT   (NROW / TILE_J)   // 32

__device__ __forceinline__ float wave_allreduce_add(float v) {
    v += __shfl_xor(v, 1, 32);
    v += __shfl_xor(v, 2, 32);
    v += __shfl_xor(v, 4, 32);
    v += __shfl_xor(v, 8, 32);
    v += __shfl_xor(v, 16, 32);
    return v;
}

__device__ __forceinline__ float fast_sigmoid(float t) {
    // v_exp_f32 + v_add + v_rcp_f32 (skip the IEEE-exact division chain)
    return __builtin_amdgcn_rcpf(1.0f + __expf(-t));
}

// B-operand fragment for one 32(K) x 16(N) tile of W^T, W torch-style [out=n][in=k].
//   lane l -> column n = nt*16 + (l&15); kbase = (l<16 ? 0 : 8);
//   element e -> K = kt*32 + kbase + (e<8 ? e : e+8)  (two contiguous 8-runs)
__device__ __forceinline__ v16bf load_w_frag(const float* __restrict__ W, int n, int kt, int lane) {
    const int kbase = kt * 32 + ((lane < 16) ? 0 : 8);
    v16bf f;
#pragma unroll
    for (int e = 0; e < 16; ++e) {
        const int k = kbase + (e < 8 ? e : e + 8);
        f[e] = (__bf16)W[n * C_DIM + k];
    }
    return f;
}

// A-operand fragment (16x32 bf16) from an LDS row-major [16][128] bf16 stage.
// Contiguous 8-half runs -> compiler fuses into ds_load_b128 pairs.
__device__ __forceinline__ v16bf load_a_frag(const __bf16* __restrict__ st, int kt, int lane) {
    const int row   = lane & 15;
    const int kbase = kt * 32 + ((lane < 16) ? 0 : 8);
    v16bf f;
#pragma unroll
    for (int e = 0; e < 16; ++e) {
        const int k = kbase + (e < 8 ? e : e + 8);
        f[e] = st[row * C_DIM + k];
    }
    return f;
}

// LN one row (128 f32, already in x4[0..3] for this lane's 4 contiguous cols),
// then write 4 packed bf16 (one ds_store_b64) into the stage.
__device__ __forceinline__ void ln_store_row(float x4[4], const float4 lnw4, const float4 lnb4,
                                             __bf16* __restrict__ stage_row, int lane) {
    float s = x4[0] + x4[1] + x4[2] + x4[3];
    s = wave_allreduce_add(s);
    const float mean = s * (1.0f / C_DIM);
    float vs = 0.f;
#pragma unroll
    for (int q = 0; q < 4; ++q) { x4[q] -= mean; vs += x4[q] * x4[q]; }
    vs = wave_allreduce_add(vs);
    const float rstd = rsqrtf(vs * (1.0f / C_DIM) + 1e-5f);
    union { __bf16 h[4]; uint2 u; } pk;
    pk.h[0] = (__bf16)(x4[0] * rstd * lnw4.x + lnb4.x);
    pk.h[1] = (__bf16)(x4[1] * rstd * lnw4.y + lnb4.y);
    pk.h[2] = (__bf16)(x4[2] * rstd * lnw4.z + lnb4.z);
    pk.h[3] = (__bf16)(x4[3] * rstd * lnw4.w + lnb4.w);
    *reinterpret_cast<uint2*>(&stage_row[4 * lane]) = pk.u;
}

// One block per row i. Fused: LN -> lin_p (+bias, optional mask) -> lin_g -> sigmoid*x -> sum_j.
__global__ __launch_bounds__(256) void branch_reduce_kernel(
    const float* __restrict__ X,                                  // [512,512,128]
    const float* __restrict__ lnw, const float* __restrict__ lnb, // [128]
    const float* __restrict__ Wp,  const float* __restrict__ bp,  // [128,128],[128]
    const float* __restrict__ Wg,  const float* __restrict__ bg,  // [128,128],[128]
    const float* __restrict__ mask,                               // [512,512] or nullptr
    float* __restrict__ zsum)                                     // [512,128]
{
    __shared__ __bf16 Astage[TILE_J * C_DIM];
    __shared__ __bf16 Xstage[TILE_J * C_DIM];
    __shared__ float  smask[TILE_J];

    const int i    = blockIdx.x;
    const int tid  = threadIdx.x;
    const int wave = tid >> 5;
    const int lane = tid & 31;
    const int n    = wave * 16 + (lane & 15);   // this wave's N column for WMMA B/C

    const float4 lnw4 = *reinterpret_cast<const float4*>(&lnw[4 * lane]);
    const float4 lnb4 = *reinterpret_cast<const float4*>(&lnb[4 * lane]);

    // Resident weight fragments: 4 K-tiles for projection and for gating.
    v16bf B1[4], B2[4];
#pragma unroll
    for (int kt = 0; kt < 4; ++kt) {
        B1[kt] = load_w_frag(Wp, n, kt, lane);
        B2[kt] = load_w_frag(Wg, n, kt, lane);
    }
    const float bias1 = bp[n];
    const float bias2 = bg[n];

    float zacc = 0.0f;

    for (int jt = 0; jt < N_JT; ++jt) {
        const int j0 = jt * TILE_J;

        // ---- LN 16 rows into bf16 A-stage (wave w handles rows w and w+8) ----
#pragma unroll
        for (int rr = 0; rr < 2; ++rr) {
            const int    row  = wave + rr * 8;
            const size_t base = ((size_t)i * NROW + (size_t)(j0 + row)) * C_DIM;
            const float4 xv   = *reinterpret_cast<const float4*>(&X[base + 4 * lane]);
            float x4[4] = {xv.x, xv.y, xv.z, xv.w};
            ln_store_row(x4, lnw4, lnb4, &Astage[row * C_DIM], lane);
        }
        if (mask && tid < TILE_J) smask[tid] = mask[(size_t)i * NROW + j0 + tid];
        if (jt + 1 < N_JT)  // hint next j-tile into cache (global_prefetch_b8)
            __builtin_prefetch(&X[((size_t)i * NROW + (size_t)(j0 + TILE_J + wave)) * C_DIM + 4 * lane], 0, 1);
        __syncthreads();

        // ---- GEMM1: x = LN(X) @ Wp^T + bp  (4 bf16 WMMA, f32 accumulate) ----
        v8f acc = {};
#pragma unroll
        for (int kt = 0; kt < 4; ++kt) {
            v16bf a = load_a_frag(Astage, kt, lane);
            acc = __builtin_amdgcn_wmma_f32_16x16x32_bf16(false, a, false, B1[kt],
                                                          (short)0, acc, false, false);
        }
        float xf[8];
#pragma unroll
        for (int r = 0; r < 8; ++r) {
            const int M = r + ((lane < 16) ? 0 : 8);
            float v = acc[r] + bias1;
            if (mask) v *= smask[M];
            xf[r] = v;
            Xstage[M * C_DIM + n] = (__bf16)v;   // restage x for the gating GEMM
        }
        __syncthreads();

        // ---- GEMM2: g = sigmoid(x @ Wg^T + bg); accumulate sum_j (g * x) ----
        v8f acc2 = {};
#pragma unroll
        for (int kt = 0; kt < 4; ++kt) {
            v16bf a = load_a_frag(Xstage, kt, lane);
            acc2 = __builtin_amdgcn_wmma_f32_16x16x32_bf16(false, a, false, B2[kt],
                                                           (short)0, acc2, false, false);
        }
        float s = 0.f;
#pragma unroll
        for (int r = 0; r < 8; ++r)
            s += fast_sigmoid(acc2[r] + bias2) * xf[r];
        s += __shfl_down(s, 16, 32);          // fold M=8..15 half onto lanes 0-15
        if (lane < 16) zacc += s;
        __syncthreads();                       // protect stages before next tile
    }

    if (lane < 16) zsum[(size_t)i * C_DIM + wave * 16 + lane] = zacc;
}

// One block per row i: out[i,j,:] = LN(z_l[i]*z_r[j]) @ Wz^T + bz
__global__ __launch_bounds__(256) void outer_proj_kernel(
    const float* __restrict__ zl,                                 // [512,128]
    const float* __restrict__ zr,                                 // [512,128]
    const float* __restrict__ lnw, const float* __restrict__ lnb, // [128]
    const float* __restrict__ Wz,  const float* __restrict__ bz,  // [128,128],[128]
    float* __restrict__ out)                                      // [512,512,128]
{
    __shared__ __bf16 Astage[TILE_J * C_DIM];

    const int i    = blockIdx.x;
    const int tid  = threadIdx.x;
    const int wave = tid >> 5;
    const int lane = tid & 31;
    const int n    = wave * 16 + (lane & 15);

    const float4 lnw4 = *reinterpret_cast<const float4*>(&lnw[4 * lane]);
    const float4 lnb4 = *reinterpret_cast<const float4*>(&lnb[4 * lane]);
    const float4 zl4  = *reinterpret_cast<const float4*>(&zl[(size_t)i * C_DIM + 4 * lane]);

    v16bf Bz[4];
#pragma unroll
    for (int kt = 0; kt < 4; ++kt) Bz[kt] = load_w_frag(Wz, n, kt, lane);
    const float biasz = bz[n];

    for (int jt = 0; jt < N_JT; ++jt) {
        const int j0 = jt * TILE_J;

#pragma unroll
        for (int rr = 0; rr < 2; ++rr) {
            const int    row  = wave + rr * 8;
            const float4 zrv  = *reinterpret_cast<const float4*>(&zr[(size_t)(j0 + row) * C_DIM + 4 * lane]);
            float x4[4] = {zl4.x * zrv.x, zl4.y * zrv.y, zl4.z * zrv.z, zl4.w * zrv.w};
            ln_store_row(x4, lnw4, lnb4, &Astage[row * C_DIM], lane);
        }
        __syncthreads();

        v8f acc = {};
#pragma unroll
        for (int kt = 0; kt < 4; ++kt) {
            v16bf a = load_a_frag(Astage, kt, lane);
            acc = __builtin_amdgcn_wmma_f32_16x16x32_bf16(false, a, false, Bz[kt],
                                                          (short)0, acc, false, false);
        }
#pragma unroll
        for (int r = 0; r < 8; ++r) {
            const int M = r + ((lane < 16) ? 0 : 8);
            out[((size_t)i * NROW + (size_t)(j0 + M)) * C_DIM + n] = acc[r] + biasz;
        }
        __syncthreads();
    }
}

extern "C" void kernel_launch(void* const* d_in, const int* in_sizes, int n_in,
                              void* d_out, int out_size, void* d_ws, size_t ws_size,
                              hipStream_t stream) {
    (void)in_sizes; (void)n_in; (void)out_size; (void)ws_size;
    const float* z      = (const float*)d_in[0];
    const float* p      = (const float*)d_in[1];
    const float* mask   = (const float*)d_in[2];
    const float* w_lp   = (const float*)d_in[3];
    const float* b_lp   = (const float*)d_in[4];
    const float* w_rp   = (const float*)d_in[5];
    const float* b_rp   = (const float*)d_in[6];
    const float* w_lg   = (const float*)d_in[7];
    const float* b_lg   = (const float*)d_in[8];
    const float* w_rg   = (const float*)d_in[9];
    const float* b_rg   = (const float*)d_in[10];
    const float* w_z    = (const float*)d_in[11];
    const float* b_z    = (const float*)d_in[12];
    const float* ln_l_w = (const float*)d_in[13];
    const float* ln_l_b = (const float*)d_in[14];
    const float* ln_r_w = (const float*)d_in[15];
    const float* ln_r_b = (const float*)d_in[16];
    const float* ln_o_w = (const float*)d_in[17];
    const float* ln_o_b = (const float*)d_in[18];

    float* zl = (float*)d_ws;             // [512,128]
    float* zr = zl + 512 * C_DIM;         // [512,128]

    // p-branch (no mask) -> z_l ; z-branch (masked) -> z_r ; then outer+proj.
    branch_reduce_kernel<<<512, 256, 0, stream>>>(p, ln_l_w, ln_l_b, w_lp, b_lp,
                                                  w_lg, b_lg, nullptr, zl);
    branch_reduce_kernel<<<512, 256, 0, stream>>>(z, ln_r_w, ln_r_b, w_rp, b_rp,
                                                  w_rg, b_rg, mask, zr);
    outer_proj_kernel<<<512, 256, 0, stream>>>(zl, zr, ln_o_w, ln_o_b, w_z, b_z,
                                               (float*)d_out);
}
